// VanillaRNN_54812372631634
// MI455X (gfx1250) — matrix operand order repaired and checked by reading, not compiled, not run on hardware
//
#include <hip/hip_runtime.h>
#include <math.h>

typedef __attribute__((ext_vector_type(16))) __bf16 v16bf;
typedef __attribute__((ext_vector_type(8)))  float  v8f;

#define HID 2048
#define SEQ 256

__device__ __forceinline__ unsigned short f2bf(float f) {
  unsigned u = __builtin_bit_cast(unsigned, f);
  u += 0x7FFFu + ((u >> 16) & 1u);          // round-to-nearest-even
  return (unsigned short)(u >> 16);
}

union BfVec { uint4 q[2]; v16bf v; };

// ---- elementwise converts -------------------------------------------------
extern "C" __global__ void __launch_bounds__(256)
cvt_f32_bf16(const float* __restrict__ src, __bf16* __restrict__ dst, int n) {
  int i = blockIdx.x * 256 + threadIdx.x;
  if (i < n) {
    unsigned short s = f2bf(src[i]);
    dst[i] = __builtin_bit_cast(__bf16, s);
  }
}

// x [2048 x 256] f32 row-major -> xT [256 x 2048] bf16 (column n contiguous in k)
extern "C" __global__ void __launch_bounds__(256)
transpose_x_bf16(const float* __restrict__ x, __bf16* __restrict__ xT) {
  int i = blockIdx.x * 256 + threadIdx.x;   // over HID*SEQ
  int m = i >> 8;                            // /SEQ
  int n = i & (SEQ - 1);
  unsigned short s = f2bf(x[i]);
  xT[(size_t)n * HID + m] = __builtin_bit_cast(__bf16, s);
}

// h1 = tanh(c + bh)  (h0 == 0, so first step needs no GEMM); transposed layout
extern "C" __global__ void __launch_bounds__(256)
h1_from_c(const float* __restrict__ cT, const float* __restrict__ bh,
          __bf16* __restrict__ hT) {
  int i = blockIdx.x * 256 + threadIdx.x;   // over SEQ*HID (n-major)
  int n = i >> 11;                           // /HID
  float v = tanhf(cT[i] + bh[n]);
  unsigned short s = f2bf(v);
  hT[i] = __builtin_bit_cast(__bf16, s);
}

// ---- fused WMMA GEMM ------------------------------------------------------
// out[M=2048, N=256] = W[2048x2048] @ B[2048x256]
//   W : bf16 row-major; BT : bf16 [N][K] (transposed B)
// mode 0: outF (f32, transposed [N][K-major=M]) = W@B            (compute cT)
// mode 1: outH (bf16, transposed) = tanh(cT + W@B + bias[n])     (RNN step)
// mode 2: outF (f32, row-major [M][N]) = W@B + bias[n]           (projection)
extern "C" __global__ void __launch_bounds__(256)
wmma_gemm(const __bf16* __restrict__ W, const __bf16* __restrict__ BT,
          const float* __restrict__ cT, const float* __restrict__ bias,
          float* __restrict__ outF, __bf16* __restrict__ outH, int mode) {
  const int lane = threadIdx.x & 31;
  const int wave = threadIdx.x >> 5;                 // 8 waves / WG
  const int m0 = blockIdx.x * 64 + (wave & 1) * 32;  // wave tile: 32(M) x 16(N)
  const int n0 = blockIdx.y * 64 + (wave >> 1) * 16;
  const int mr = lane & 15;
  const int kh = lane >> 4;

  const __bf16* a0p = W  + (size_t)(m0 + mr) * HID + kh * 8;   // A rows m0..m0+15
  const __bf16* a1p = a0p + (size_t)16 * HID;                  // A rows m0+16..m0+31
  const __bf16* btp = BT + (size_t)(n0 + mr) * HID + kh * 8;   // B col n0+mr

  v8f acc0 = {};
  v8f acc1 = {};

  for (int k = 0; k < HID; k += 32) {
    BfVec a0, a1, b;
    a0.q[0] = *(const uint4*)(a0p + k);        // K = k+8kh   .. +7  -> V0-3
    a0.q[1] = *(const uint4*)(a0p + k + 16);   // K = k+16+8kh.. +7  -> V4-7
    a1.q[0] = *(const uint4*)(a1p + k);
    a1.q[1] = *(const uint4*)(a1p + k + 16);
    b.q[0]  = *(const uint4*)(btp + k);        // K = k+8kh   .. +7  -> V0-3
    b.q[1]  = *(const uint4*)(btp + k + 16);   // K = k+16+8kh.. +7  -> V4-7
    acc0 = __builtin_amdgcn_wmma_f32_16x16x32_bf16(false, a0.v, false, b.v,
                                                   (short)0, acc0, false, false);
    acc1 = __builtin_amdgcn_wmma_f32_16x16x32_bf16(false, a1.v, false, b.v,
                                                   (short)0, acc1, false, false);
  }

  // C/D layout: VGPR r, lanes 0-15 -> (M=r,    N=lane), lanes 16-31 -> (M=r+8)
  const int n  = n0 + mr;
  const int mb = kh * 8;
  const float bia = bias ? bias[n] : 0.0f;
#pragma unroll
  for (int t = 0; t < 2; ++t) {
    v8f acc = t ? acc1 : acc0;
    const int m = m0 + t * 16 + mb;            // 8 contiguous M rows per lane
    if (mode == 0) {
      float* dst = outF + (size_t)n * HID + m;
#pragma unroll
      for (int r = 0; r < 8; ++r) dst[r] = acc[r];
    } else if (mode == 1) {
      const float* cp = cT + (size_t)n * HID + m;
      unsigned int w[4];
#pragma unroll
      for (int r = 0; r < 4; ++r) {
        unsigned short lo = f2bf(tanhf(cp[2 * r]     + acc[2 * r]     + bia));
        unsigned short hi = f2bf(tanhf(cp[2 * r + 1] + acc[2 * r + 1] + bia));
        w[r] = (unsigned)lo | ((unsigned)hi << 16);
      }
      *(uint4*)(outH + (size_t)n * HID + m) = make_uint4(w[0], w[1], w[2], w[3]);
    } else {
#pragma unroll
      for (int r = 0; r < 8; ++r)
        outF[(size_t)(m + r) * SEQ + n] = acc[r] + bia;
    }
  }
}

// ---- row softmax over 256 columns ----------------------------------------
extern "C" __global__ void __launch_bounds__(256)
softmax_rows(const float* __restrict__ p, float* __restrict__ out) {
  __shared__ float sm[256];
  const int t = threadIdx.x;
  const int row = blockIdx.x;
  float v = p[(size_t)row * SEQ + t];
  sm[t] = v; __syncthreads();
  for (int s = 128; s > 0; s >>= 1) {
    if (t < s) sm[t] = fmaxf(sm[t], sm[t + s]);
    __syncthreads();
  }
  float mx = sm[0]; __syncthreads();
  float e = expf(v - mx);
  sm[t] = e; __syncthreads();
  for (int s = 128; s > 0; s >>= 1) {
    if (t < s) sm[t] += sm[t + s];
    __syncthreads();
  }
  out[(size_t)row * SEQ + t] = e / sm[0];
}

// ---- host launch ----------------------------------------------------------
extern "C" void kernel_launch(void* const* d_in, const int* in_sizes, int n_in,
                              void* d_out, int out_size, void* d_ws, size_t ws_size,
                              hipStream_t stream) {
  (void)in_sizes; (void)n_in; (void)out_size; (void)ws_size;
  const float* x   = (const float*)d_in[0];
  const float* Whx = (const float*)d_in[1];
  const float* Whh = (const float*)d_in[2];
  const float* Wph = (const float*)d_in[3];
  const float* bh  = (const float*)d_in[4];
  const float* bp  = (const float*)d_in[5];
  float* out = (float*)d_out;

  char* ws = (char*)d_ws;
  const size_t WSZ  = (size_t)HID * HID * 2;     // 8 MB per bf16 weight
  const size_t HSZB = (size_t)SEQ * HID * 2;     // 1 MB bf16 state
  const size_t HSZF = (size_t)SEQ * HID * 4;     // 2 MB f32 state
  __bf16* Whh_b = (__bf16*)(ws);
  __bf16* Whx_b = (__bf16*)(ws + WSZ);
  __bf16* Wph_b = (__bf16*)(ws + 2 * WSZ);
  __bf16* xT    = (__bf16*)(ws + 3 * WSZ);
  float*  cT    = (float*)(ws + 3 * WSZ + HSZB);
  __bf16* hA    = (__bf16*)((char*)cT + HSZF);
  __bf16* hB    = (__bf16*)((char*)hA + HSZB);
  float*  P     = (float*)((char*)hB + HSZB);

  dim3 blk(256);
  const int nW = HID * HID;
  cvt_f32_bf16<<<nW / 256, blk, 0, stream>>>(Whh, Whh_b, nW);
  cvt_f32_bf16<<<nW / 256, blk, 0, stream>>>(Whx, Whx_b, nW);
  cvt_f32_bf16<<<nW / 256, blk, 0, stream>>>(Wph, Wph_b, nW);
  transpose_x_bf16<<<(HID * SEQ) / 256, blk, 0, stream>>>(x, xT);

  dim3 gg(HID / 64, SEQ / 64);   // 32 x 4 workgroups, 8 waves each
  // cT = Whx @ x
  wmma_gemm<<<gg, blk, 0, stream>>>(Whx_b, xT, nullptr, nullptr, cT, nullptr, 0);
  // h1 = tanh(c + bh)   (h0 == 0)
  h1_from_c<<<(HID * SEQ) / 256, blk, 0, stream>>>(cT, bh, hA);

  // steps 2..256: h <- tanh(c + Whh@h + bh)
  __bf16* cur = hA; __bf16* nxt = hB;
  for (int i = 0; i < SEQ - 1; ++i) {
    wmma_gemm<<<gg, blk, 0, stream>>>(Whh_b, cur, cT, bh, nullptr, nxt, 1);
    __bf16* tmp = cur; cur = nxt; nxt = tmp;
  }

  // p = Wph @ hS + bp ; softmax rows
  wmma_gemm<<<gg, blk, 0, stream>>>(Wph_b, cur, nullptr, bp, P, nullptr, 2);
  softmax_rows<<<HID, blk, 0, stream>>>(P, out);
}
